// GridSelfAttentionCpp_62182536511491
// MI455X (gfx1250) — compile-verified, matrix operand order checked by reading, not run on hardware
//
#include <hip/hip_runtime.h>
#include <hip/hip_bf16.h>

// ---------------------------------------------------------------------------
// Grid self-attention (triangle attention), N=384, C=128, H=4, DH=32.
// All matmuls use v_wmma_f32_16x16x32_f16 (CDNA5 wave32 WMMA).
// ---------------------------------------------------------------------------

#define NRES 384
#define CPAIR 128
#define NHEAD 4
#define DHEAD 32
#define NPOS (NRES * NRES)          // 147456 pair positions
#define LN_EPS 1e-5f
#define NEG_INF -1e9f

typedef __attribute__((ext_vector_type(16))) _Float16 v16h;
typedef __attribute__((ext_vector_type(8)))  _Float16 v8h;
typedef __attribute__((ext_vector_type(4)))  _Float16 v4h;
typedef __attribute__((ext_vector_type(8)))  float    v8f;

__device__ __forceinline__ v16h cat16(v8h lo, v8h hi) {
    return __builtin_shufflevector(lo, hi, 0,1,2,3,4,5,6,7,8,9,10,11,12,13,14,15);
}

// ---------------------------------------------------------------------------
// Kernel 1: LayerNorm over C=128 + fused per-head pair bias.
// One wave (32 lanes) per pair position; 4 channels per lane.
//   xh[p*128+c]           = LN(pair)[p,c]  (f16)
//   biasw[h*NPOS + p]     = sum_c x[p,c] * w_bias[c,h]  (f32)
// ---------------------------------------------------------------------------
__global__ __launch_bounds__(256) void ln_bias_kernel(
    const float* __restrict__ pair, const float* __restrict__ ln_w,
    const float* __restrict__ ln_b, const float* __restrict__ w_bias,
    _Float16* __restrict__ xh, float* __restrict__ biasw)
{
    const int wave = threadIdx.x >> 5;
    const int lane = threadIdx.x & 31;
    const long p = (long)blockIdx.x * 8 + wave;

    const float4 xv = *(const float4*)(pair + p * CPAIR + lane * 4);

    float s = xv.x + xv.y + xv.z + xv.w;
    s += __shfl_xor(s, 1, 32);  s += __shfl_xor(s, 2, 32);
    s += __shfl_xor(s, 4, 32);  s += __shfl_xor(s, 8, 32);
    s += __shfl_xor(s, 16, 32);
    const float mu = s * (1.0f / 128.0f);

    const float d0 = xv.x - mu, d1 = xv.y - mu, d2 = xv.z - mu, d3 = xv.w - mu;
    float vs = d0 * d0 + d1 * d1 + d2 * d2 + d3 * d3;
    vs += __shfl_xor(vs, 1, 32);  vs += __shfl_xor(vs, 2, 32);
    vs += __shfl_xor(vs, 4, 32);  vs += __shfl_xor(vs, 8, 32);
    vs += __shfl_xor(vs, 16, 32);
    const float rs = rsqrtf(vs * (1.0f / 128.0f) + LN_EPS);

    const float4 w4 = *(const float4*)(ln_w + lane * 4);
    const float4 b4 = *(const float4*)(ln_b + lane * 4);
    float y[4];
    y[0] = d0 * rs * w4.x + b4.x;
    y[1] = d1 * rs * w4.y + b4.y;
    y[2] = d2 * rs * w4.z + b4.z;
    y[3] = d3 * rs * w4.w + b4.w;

    v4h o;
    o[0] = (_Float16)y[0]; o[1] = (_Float16)y[1];
    o[2] = (_Float16)y[2]; o[3] = (_Float16)y[3];
    *(v4h*)(xh + p * CPAIR + lane * 4) = o;

    // fused pair-bias projection: w_bias is [C,H] row-major, H=4.
    float hb0 = 0.f, hb1 = 0.f, hb2 = 0.f, hb3 = 0.f;
#pragma unroll
    for (int j = 0; j < 4; ++j) {
        const float4 wb = *(const float4*)(w_bias + (lane * 4 + j) * NHEAD);
        hb0 += y[j] * wb.x;  hb1 += y[j] * wb.y;
        hb2 += y[j] * wb.z;  hb3 += y[j] * wb.w;
    }
#pragma unroll
    for (int m = 1; m < 32; m <<= 1) {
        hb0 += __shfl_xor(hb0, m, 32);  hb1 += __shfl_xor(hb1, m, 32);
        hb2 += __shfl_xor(hb2, m, 32);  hb3 += __shfl_xor(hb3, m, 32);
    }
    if (lane == 0) {
        biasw[0L * NPOS + p] = hb0;
        biasw[1L * NPOS + p] = hb1;
        biasw[2L * NPOS + p] = hb2;
        biasw[3L * NPOS + p] = hb3;
    }
}

// ---------------------------------------------------------------------------
// Kernel 2: generic D = A[M,128] @ W[128,128] via WMMA f16.
// 8 waves/block; each wave computes a 16x128 strip (8 accumulator tiles).
// Weights staged transposed in LDS as Wt[n][k] f16 so the B fragment is two
// contiguous 16B loads per lane (matches 16-bit B VGPR layout).
// modes: 0 = f16 store, 1 = sigmoid f16, 2 = transposed-V f16 store
//        ([b,h,dh,k] layout for contiguous PV B fragments), 3 = f32 store.
// ---------------------------------------------------------------------------
__global__ __launch_bounds__(256) void gemm128_kernel(
    const _Float16* __restrict__ A, const float* __restrict__ W,
    void* __restrict__ Dout, int mode)
{
    __shared__ __align__(16) _Float16 Wt[128 * 128];
    for (int i = threadIdx.x; i < 128 * 128; i += 256) {
        const int c = i >> 7, d = i & 127;
        Wt[d * 128 + c] = (_Float16)W[i];
    }
    __syncthreads();

    const int wave = threadIdx.x >> 5;
    const int lane = threadIdx.x & 31;
    const int half = lane >> 4;        // which 16-lane half
    const int ln   = lane & 15;

    const long row0 = (long)blockIdx.x * 128 + wave * 16;
    const long arow = row0 + ln;

    v8f acc[8];
    const v8f zero = {};
#pragma unroll
    for (int i = 0; i < 8; ++i) acc[i] = zero;

#pragma unroll
    for (int kk = 0; kk < 4; ++kk) {
        const _Float16* ap = A + arow * CPAIR + kk * 32 + half * 8;
        const v16h afrag = cat16(*(const v8h*)ap, *(const v8h*)(ap + 16));
#pragma unroll
        for (int ct = 0; ct < 8; ++ct) {
            const _Float16* bp = Wt + (ct * 16 + ln) * 128 + kk * 32 + half * 8;
            const v16h bfrag = cat16(*(const v8h*)bp, *(const v8h*)(bp + 16));
            acc[ct] = __builtin_amdgcn_wmma_f32_16x16x32_f16(
                false, afrag, false, bfrag, (short)0, acc[ct], false, false);
        }
    }

#pragma unroll
    for (int ct = 0; ct < 8; ++ct) {
#pragma unroll
        for (int r = 0; r < 8; ++r) {
            const long m = row0 + half * 8 + r;   // output row (C/D layout)
            const int col = ct * 16 + ln;          // output col
            const float vf = acc[ct][r];
            if (mode == 0) {
                ((_Float16*)Dout)[m * CPAIR + col] = (_Float16)vf;
            } else if (mode == 1) {
                ((_Float16*)Dout)[m * CPAIR + col] =
                    (_Float16)(1.0f / (1.0f + __expf(-vf)));
            } else if (mode == 3) {
                ((float*)Dout)[m * CPAIR + col] = vf;
            } else {  // mode 2: V transposed to [b, h, dh, k]
                const int b = (int)(m / NRES), key = (int)(m % NRES);
                const int h = col >> 5, dh = col & 31;
                ((_Float16*)Dout)[(((long)(b * NHEAD + h) * DHEAD + dh) * NRES) + key] =
                    (_Float16)vf;
            }
        }
    }
}

// ---------------------------------------------------------------------------
// Kernel 3: flash attention per (b, h). Each wave owns 16 query rows; block of
// 8 waves covers 128 rows; grid = (3, H, N). Streaming over keys in chunks of
// 32 with online softmax. Probabilities are staged through per-wave LDS to
// convert the WMMA D layout into the A layout for the PV matmul.
// ---------------------------------------------------------------------------
__global__ __launch_bounds__(256) void attn_kernel(
    const _Float16* __restrict__ qh, const _Float16* __restrict__ kh,
    const _Float16* __restrict__ vT, const float* __restrict__ biasw,
    const int* __restrict__ mask, const _Float16* __restrict__ gate,
    _Float16* __restrict__ og)
{
    __shared__ __align__(16) _Float16 pstage[8][16 * 32];

    const int wave = threadIdx.x >> 5;
    const int lane = threadIdx.x & 31;
    const int half = lane >> 4;
    const int ln   = lane & 15;

    const int b  = blockIdx.z;
    const int h  = blockIdx.y;
    const int q0 = blockIdx.x * 128 + wave * 16;
    const float scale = 0.17677669529663687f;  // 1/sqrt(32)

    // Q fragment (A layout): row = q0+ln, dh split per ISA 16-bit A layout.
    const _Float16* qp = qh + ((long)(b * NRES + q0 + ln) * CPAIR + h * DHEAD + half * 8);
    const v16h qf = cat16(*(const v8h*)qp, *(const v8h*)(qp + 16));

    v8f o0 = {}, o1 = {};
    float rmax[8], lsum[8];
#pragma unroll
    for (int r = 0; r < 8; ++r) { rmax[r] = -1e30f; lsum[r] = 0.0f; }

    for (int kk = 0; kk < NRES / 32; ++kk) {
        const int kbase = kk * 32;

        // S = Q @ K^T for two 16-key tiles (K rows are contiguous in dh).
        v8f s[2];
#pragma unroll
        for (int t = 0; t < 2; ++t) {
            const int key = kbase + t * 16 + ln;
            const _Float16* kp =
                kh + ((long)(b * NRES + key) * CPAIR + h * DHEAD + half * 8);
            const v16h kf = cat16(*(const v8h*)kp, *(const v8h*)(kp + 16));
            const v8f z = {};
            s[t] = __builtin_amdgcn_wmma_f32_16x16x32_f16(
                false, qf, false, kf, (short)0, z, false, false);
        }

        // scale + pair bias + key mask (D layout: lane owns col=key, 8 rows).
#pragma unroll
        for (int t = 0; t < 2; ++t) {
            const int key = kbase + t * 16 + ln;
            const int mk = mask[b * NRES + key];
            const float* bp =
                biasw + ((long)h * NRES + q0 + half * 8) * NRES + key;
#pragma unroll
            for (int r = 0; r < 8; ++r) {
                const float bv = bp[(long)r * NRES];
                s[t][r] = mk ? (s[t][r] * scale + bv) : NEG_INF;
            }
        }

        // Online softmax: row stats live in 16-lane halves (rows 0-7 in lanes
        // 0-15, rows 8-15 in lanes 16-31) matching the C/D VGPR layout.
#pragma unroll
        for (int r = 0; r < 8; ++r) {
            float tm = fmaxf(s[0][r], s[1][r]);
            tm = fmaxf(tm, __shfl_xor(tm, 1, 32));
            tm = fmaxf(tm, __shfl_xor(tm, 2, 32));
            tm = fmaxf(tm, __shfl_xor(tm, 4, 32));
            tm = fmaxf(tm, __shfl_xor(tm, 8, 32));
            const float nm   = fmaxf(rmax[r], tm);
            const float corr = __expf(rmax[r] - nm);
            rmax[r] = nm;
            lsum[r] *= corr;
            o0[r] *= corr;
            o1[r] *= corr;
            const float p0 = __expf(s[0][r] - nm);
            const float p1 = __expf(s[1][r] - nm);
            lsum[r] += p0 + p1;
            const int mrow = half * 8 + r;
            pstage[wave][mrow * 32 + ln]      = (_Float16)p0;
            pstage[wave][mrow * 32 + 16 + ln] = (_Float16)p1;
        }

        // Re-load P as an A fragment (wave-private LDS; DScnt ordering only).
        const _Float16* pp = &pstage[wave][ln * 32 + half * 8];
        const v16h pf = cat16(*(const v8h*)pp, *(const v8h*)(pp + 16));

        // O += P @ V : vT[b,h,dh,k] gives contiguous key runs per lane.
#pragma unroll
        for (int t = 0; t < 2; ++t) {
            const _Float16* vp =
                vT + (((long)(b * NHEAD + h) * DHEAD + t * 16 + ln) * NRES +
                      kbase + half * 8);
            const v16h vf = cat16(*(const v8h*)vp, *(const v8h*)(vp + 16));
            if (t == 0)
                o0 = __builtin_amdgcn_wmma_f32_16x16x32_f16(
                    false, pf, false, vf, (short)0, o0, false, false);
            else
                o1 = __builtin_amdgcn_wmma_f32_16x16x32_f16(
                    false, pf, false, vf, (short)0, o1, false, false);
        }
    }

    // Normalize, gate, store og (f16).
#pragma unroll
    for (int r = 0; r < 8; ++r) {
        float ls = lsum[r];
        ls += __shfl_xor(ls, 1, 32);
        ls += __shfl_xor(ls, 2, 32);
        ls += __shfl_xor(ls, 4, 32);
        ls += __shfl_xor(ls, 8, 32);
        const float inv = 1.0f / ls;
        const int q = q0 + half * 8 + r;
        const long base = (long)(b * NRES + q) * CPAIR + h * DHEAD;
        const float g0 = (float)gate[base + ln];
        const float g1 = (float)gate[base + 16 + ln];
        og[base + ln]      = (_Float16)(o0[r] * inv * g0);
        og[base + 16 + ln] = (_Float16)(o1[r] * inv * g1);
    }
}

// ---------------------------------------------------------------------------
// Launch: LN+bias -> 4 projections -> flash attention -> output projection.
// Workspace layout (bytes, 256-aligned):
//   xh   f16 [NPOS,128]      @ 0
//   qh   f16 [NPOS,128]      @ 37748736
//   kh   f16 [NPOS,128]      @ 75497472
//   vT   f16 [N,H,DH,N]      @ 113246208
//   gate f16 [NPOS,128]      @ 150994944
//   og   f16 [NPOS,128]      @ 188743680
//   bias f32 [H,N,N]         @ 226492416   (end ~228.9 MB)
// ---------------------------------------------------------------------------
extern "C" void kernel_launch(void* const* d_in, const int* in_sizes, int n_in,
                              void* d_out, int out_size, void* d_ws, size_t ws_size,
                              hipStream_t stream) {
    (void)in_sizes; (void)n_in; (void)out_size; (void)ws_size;

    const float* pair   = (const float*)d_in[0];
    const int*   mask   = (const int*)d_in[1];
    const float* ln_w   = (const float*)d_in[2];
    const float* ln_b   = (const float*)d_in[3];
    const float* w_bias = (const float*)d_in[4];
    const float* wq     = (const float*)d_in[5];
    const float* wk     = (const float*)d_in[6];
    const float* wv     = (const float*)d_in[7];
    const float* wg     = (const float*)d_in[8];
    const float* wo     = (const float*)d_in[9];

    char* ws = (char*)d_ws;
    const size_t SZ_F16 = (size_t)NPOS * CPAIR * sizeof(_Float16);  // 37748736
    _Float16* xh   = (_Float16*)(ws);
    _Float16* qh   = (_Float16*)(ws + 1 * SZ_F16);
    _Float16* kh   = (_Float16*)(ws + 2 * SZ_F16);
    _Float16* vT   = (_Float16*)(ws + 3 * SZ_F16);
    _Float16* gate = (_Float16*)(ws + 4 * SZ_F16);
    _Float16* og   = (_Float16*)(ws + 5 * SZ_F16);
    float*    bw   = (float*)(ws + 6 * SZ_F16);

    ln_bias_kernel<<<NPOS / 8, 256, 0, stream>>>(pair, ln_w, ln_b, w_bias, xh, bw);

    const int gblocks = NPOS / 128;  // 1152
    gemm128_kernel<<<gblocks, 256, 0, stream>>>(xh, wq, qh, 0);
    gemm128_kernel<<<gblocks, 256, 0, stream>>>(xh, wk, kh, 0);
    gemm128_kernel<<<gblocks, 256, 0, stream>>>(xh, wv, vT, 2);
    gemm128_kernel<<<gblocks, 256, 0, stream>>>(xh, wg, gate, 1);

    dim3 agrid(NRES / 128, NHEAD, NRES);  // (3, 4, 384)
    attn_kernel<<<agrid, 256, 0, stream>>>(qh, kh, vT, bw, mask, gate, og);

    gemm128_kernel<<<gblocks, 256, 0, stream>>>(og, wo, d_out, 3);
}
